// TransformerBlock_81518479278631
// MI455X (gfx1250) — compile-verified, matrix operand order checked by reading, not run on hardware
//
#include <hip/hip_runtime.h>
#include <hip/hip_bf16.h>
#include <math.h>

// ---- problem constants (match reference) ----
#define D_MODEL   1024
#define NUM_HEADS 16
#define HEAD_DIM  64
#define MLP_DIM   4096
#define BATCH     2
#define SEQ       2048
#define ROWS      (BATCH * SEQ)   // 4096
#define LN_EPS    1e-5f

typedef __attribute__((ext_vector_type(16))) __bf16 v16bf;
typedef __attribute__((ext_vector_type(8)))  __bf16 v8bf;
typedef __attribute__((ext_vector_type(8)))  float  v8f;
typedef __attribute__((ext_vector_type(4)))  __bf16 v4bf;
typedef int v4i __attribute__((vector_size(16)));   // matches builtin param type

// ---- CDNA5 async global->LDS path (guarded; falls back to vector copies) ----
#if defined(__AMDGCN__) && __has_builtin(__builtin_amdgcn_global_load_async_to_lds_b128)
#define HAS_ASYNC_LDS 1
#else
#define HAS_ASYNC_LDS 0
#endif

#if HAS_ASYNC_LDS
#if __has_builtin(__builtin_amdgcn_s_wait_asynccnt)
#define WAIT_ASYNC(n) __builtin_amdgcn_s_wait_asynccnt(n)
#else
#define WAIT_ASYNC(n) asm volatile("s_wait_asynccnt %0" ::"i"(n) : "memory")
#endif
#else
#define WAIT_ASYNC(n) do { } while (0)
#endif

static __device__ __forceinline__ void async_cp16(const __bf16* g, __bf16* l) {
#if HAS_ASYNC_LDS
    // global_load_async_to_lds_b128: 16B global -> LDS, tracked by ASYNCcnt
    __builtin_amdgcn_global_load_async_to_lds_b128((v4i*)g, (v4i*)l, 0, 0);
#else
    *reinterpret_cast<v8bf*>(l) = *reinterpret_cast<const v8bf*>(g);
#endif
}

static __device__ __forceinline__ v8f wmma_bf16(v16bf a, v16bf b, v8f c) {
    // D = A(16x32 bf16) * B(32x16 bf16) + C(16x16 f32)
    return __builtin_amdgcn_wmma_f32_16x16x32_bf16(
        /*neg_a=*/false, a, /*neg_b=*/false, b,
        /*c_mod=*/(short)0, c, /*reuse_a=*/false, /*reuse_b=*/false);
}

static __device__ __forceinline__ float gelu_tanh(float x) {
    const float c0 = 0.7978845608028654f;   // sqrt(2/pi)
    return 0.5f * x * (1.0f + tanhf(c0 * (x + 0.044715f * x * x * x)));
}

// ---------------------------------------------------------------------------
// fp32 -> bf16 conversion (weights / activations)
// ---------------------------------------------------------------------------
__global__ void cvt_f32_bf16_kernel(const float* __restrict__ src,
                                    __bf16* __restrict__ dst, int n) {
    int i = blockIdx.x * blockDim.x + threadIdx.x;
    int stride = gridDim.x * blockDim.x;
    for (; i < n; i += stride) dst[i] = (__bf16)src[i];
}

// ---------------------------------------------------------------------------
// LayerNorm over last dim (C = D_MODEL), one 256-thread block per row,
// output bf16 (feeds WMMA GEMMs)
// ---------------------------------------------------------------------------
__global__ void ln_kernel(const float* __restrict__ x,
                          const float* __restrict__ g,
                          const float* __restrict__ b,
                          __bf16* __restrict__ out) {
    const int row = blockIdx.x;
    const float* xr = x + (size_t)row * D_MODEL;
    float s = 0.f, ss = 0.f;
    for (int i = threadIdx.x; i < D_MODEL; i += blockDim.x) {
        float v = xr[i];
        s += v;
        ss += v * v;
    }
    __shared__ float sh[2][8];
    const int lane = threadIdx.x & 31;
    const int wave = threadIdx.x >> 5;
#pragma unroll
    for (int o = 16; o > 0; o >>= 1) {
        s  += __shfl_down(s, o);
        ss += __shfl_down(ss, o);
    }
    if (lane == 0) { sh[0][wave] = s; sh[1][wave] = ss; }
    __syncthreads();
    s = 0.f; ss = 0.f;
#pragma unroll
    for (int w = 0; w < 8; ++w) { s += sh[0][w]; ss += sh[1][w]; }
    const float mu   = s  * (1.0f / D_MODEL);
    const float var  = ss * (1.0f / D_MODEL) - mu * mu;
    const float rstd = rsqrtf(var + LN_EPS);
    __bf16* orow = out + (size_t)row * D_MODEL;
    for (int i = threadIdx.x; i < D_MODEL; i += blockDim.x)
        orow[i] = (__bf16)((xr[i] - mu) * rstd * g[i] + b[i]);
}

// ---------------------------------------------------------------------------
// bf16 WMMA GEMM:  C[M,N] = A[M,K] @ B[K,N] + bias (+ res)
// 256 threads = 8 waves; block tile 64(M) x 128(N); K step 32.
// Wave tile 32x32 (2x2 WMMAs). LDS double-buffered via async global->LDS.
// ACT: 0 = none, 1 = tanh-GELU (applied before residual add).
// ---------------------------------------------------------------------------
#define BM 64
#define BN 128
#define BK 32

template <int ACT>
__global__ __launch_bounds__(256)
void gemm_bf16_kernel(const __bf16* __restrict__ A, const __bf16* __restrict__ Bw,
                      const float* __restrict__ bias, const float* __restrict__ res,
                      float* __restrict__ outF, __bf16* __restrict__ outB,
                      int M, int N, int K) {
    __shared__ __bf16 As[2][BM][BK];   // 2 * 4 KB
    __shared__ __bf16 Bs[2][BK][BN];   // 2 * 8 KB
    const int tid  = threadIdx.x;
    const int lane = tid & 31;
    const int wave = tid >> 5;
    const int wm = (wave >> 2) * 32;   // 0 / 32
    const int wn = (wave & 3) * 32;    // 0,32,64,96
    const int bm = blockIdx.y * BM;
    const int bn = blockIdx.x * BN;
    const int half = lane >> 4;        // K-half selector
    const int mr   = lane & 15;        // A row / B col within 16x16 tile

    // staging indices (per thread: 1x16B for A, 2x16B for B)
    const int ar = tid >> 2;           // 0..63
    const int ac = (tid & 3) * 8;      // 0,8,16,24
    const int br = tid >> 3;           // 0..31
    const int bc = (tid & 7) * 16;     // 0..112

    v8f acc[2][2] = {};
    const int nk = K / BK;

    // prologue: stage K-step 0 into buffer 0
    {
        const __bf16* ga = &A[(size_t)(bm + ar) * K + ac];
        const __bf16* gb = &Bw[(size_t)br * N + (bn + bc)];
        async_cp16(ga,     &As[0][ar][ac]);
        async_cp16(gb,     &Bs[0][br][bc]);
        async_cp16(gb + 8, &Bs[0][br][bc + 8]);
    }

    for (int ks = 0; ks < nk; ++ks) {
        const int cur = ks & 1;
        if (ks + 1 < nk) {
            const int nxt = cur ^ 1;
            const int k0 = (ks + 1) * BK;
            const __bf16* ga = &A[(size_t)(bm + ar) * K + (k0 + ac)];
            const __bf16* gb = &Bw[(size_t)(k0 + br) * N + (bn + bc)];
            async_cp16(ga,     &As[nxt][ar][ac]);
            async_cp16(gb,     &Bs[nxt][br][bc]);
            async_cp16(gb + 8, &Bs[nxt][br][bc + 8]);
            WAIT_ASYNC(3);   // newest 3 may remain outstanding -> prev panel done
        } else {
            WAIT_ASYNC(0);
        }
        __syncthreads();

        v16bf af[2], bfr[2];
#pragma unroll
        for (int i = 0; i < 2; ++i)
#pragma unroll
            for (int g2 = 0; g2 < 2; ++g2)
#pragma unroll
                for (int j = 0; j < 8; ++j) {
                    const int kk = g2 * 16 + half * 8 + j;
                    af[i][g2 * 8 + j]  = As[cur][wm + i * 16 + mr][kk];
                    bfr[i][g2 * 8 + j] = Bs[cur][kk][wn + i * 16 + mr];
                }
#pragma unroll
        for (int i = 0; i < 2; ++i)
#pragma unroll
            for (int jn = 0; jn < 2; ++jn)
                acc[i][jn] = wmma_bf16(af[i], bfr[jn], acc[i][jn]);
        __syncthreads();
    }

    // epilogue
#pragma unroll
    for (int jn = 0; jn < 2; ++jn) {
        const int n = bn + wn + jn * 16 + mr;
        const float bvv = bias ? bias[n] : 0.f;
#pragma unroll
        for (int i = 0; i < 2; ++i) {
            const int mbase = bm + wm + i * 16 + half * 8;
#pragma unroll
            for (int r = 0; r < 8; ++r) {
                const int m = mbase + r;
                float v = acc[i][jn][r] + bvv;
                if (ACT == 1) v = gelu_tanh(v);
                if (res)  v += res[(size_t)m * N + n];
                if (outF) outF[(size_t)m * N + n] = v;
                if (outB) outB[(size_t)m * N + n] = (__bf16)v;
            }
        }
    }
}

// ---------------------------------------------------------------------------
// Causal flash attention, one wave per (b, h, 16-row Q tile).
// Q/K/V/O layout: [B, T, H*Dh] bf16. Online softmax; QK^T and P*V via WMMA.
// No block barriers (waves have divergent causal trip counts).
// ---------------------------------------------------------------------------
__global__ __launch_bounds__(128)
void attn_kernel(const __bf16* __restrict__ Q, const __bf16* __restrict__ Km,
                 const __bf16* __restrict__ V, __bf16* __restrict__ O) {
    __shared__ float Pl[4][16][32];   // per-wave P bounce (C-layout -> A-layout)
    const int lane = threadIdx.x & 31;
    const int wave = threadIdx.x >> 5;
    const int work = blockIdx.x * 4 + wave;
    const int QT = SEQ / 16;                       // 128 q-tiles
    const int qt = work % QT;
    const int h  = (work / QT) % NUM_HEADS;
    const int b  = work / (QT * NUM_HEADS);
    const size_t base = ((size_t)b * SEQ) * D_MODEL + (size_t)h * HEAD_DIM;
    const int qbase = qt * 16;
    const int half = lane >> 4;
    const int mr   = lane & 15;
    const float scale = 0.125f;                    // 1/sqrt(64)

    // Q A-fragments for contraction halves d=[0,32) and [32,64)
    v16bf qf[2];
#pragma unroll
    for (int t = 0; t < 2; ++t)
#pragma unroll
        for (int g2 = 0; g2 < 2; ++g2)
#pragma unroll
            for (int j = 0; j < 8; ++j)
                qf[t][g2 * 8 + j] =
                    Q[base + (size_t)(qbase + mr) * D_MODEL +
                      (t * 32 + g2 * 16 + half * 8 + j)];

    v8f acc[4] = {};                // 16 x 64 output accumulator (f32)
    float rm[8], rl[8];
#pragma unroll
    for (int r = 0; r < 8; ++r) { rm[r] = -3.0e38f; rl[r] = 0.f; }

    const int nblk = (qt + 2) >> 1;       // ceil((qbase+16)/32)
    for (int kb = 0; kb < nblk; ++kb) {
        const int kvbase = kb * 32;
        if (kb + 1 < nblk) {               // prefetch next K/V rows into L2/WGP$
            __builtin_prefetch(&Km[base + (size_t)(kvbase + 32 + mr) * D_MODEL], 0, 1);
            __builtin_prefetch(&V[base + (size_t)(kvbase + 32 + mr) * D_MODEL], 0, 1);
        }
        // ---- scores S = Q K^T : two 16x16 tiles over 32 kv positions ----
        v8f s[2] = {};
#pragma unroll
        for (int t = 0; t < 2; ++t) {
#pragma unroll
            for (int kd = 0; kd < 2; ++kd) {
                v16bf kfrag;
#pragma unroll
                for (int g2 = 0; g2 < 2; ++g2)
#pragma unroll
                    for (int j = 0; j < 8; ++j)
                        kfrag[g2 * 8 + j] =
                            Km[base + (size_t)(kvbase + t * 16 + mr) * D_MODEL +
                               (kd * 32 + g2 * 16 + half * 8 + j)];
                s[t] = wmma_bf16(qf[kd], kfrag, s[t]);
            }
        }
        // ---- causal mask + scale ----
        const int qrow_base = qbase + half * 8;
#pragma unroll
        for (int t = 0; t < 2; ++t) {
            const int kv = kvbase + t * 16 + mr;
#pragma unroll
            for (int r = 0; r < 8; ++r) {
                float v = s[t][r] * scale;
                if (kv > qrow_base + r) v = -3.0e38f;
                s[t][r] = v;
            }
        }
        // ---- online softmax (row reductions across 16-lane groups) ----
#pragma unroll
        for (int r = 0; r < 8; ++r) {
            float mblk = fmaxf(s[0][r], s[1][r]);
#pragma unroll
            for (int o = 8; o > 0; o >>= 1)
                mblk = fmaxf(mblk, __shfl_xor(mblk, o));
            const float mnew = fmaxf(rm[r], mblk);
            const float corr = __expf(rm[r] - mnew);
            const float p0 = __expf(s[0][r] - mnew);
            const float p1 = __expf(s[1][r] - mnew);
            float ps = p0 + p1;
#pragma unroll
            for (int o = 8; o > 0; o >>= 1) ps += __shfl_xor(ps, o);
            rl[r] = rl[r] * corr + ps;
            rm[r] = mnew;
#pragma unroll
            for (int dt = 0; dt < 4; ++dt) acc[dt][r] *= corr;
            const int m = half * 8 + r;
            Pl[wave][m][mr]      = p0;    // C-layout -> LDS (wave-private)
            Pl[wave][m][16 + mr] = p1;
        }
        // ---- P (16x32) as A-fragment (bf16), then P @ V ----
        v16bf pa;
#pragma unroll
        for (int g2 = 0; g2 < 2; ++g2)
#pragma unroll
            for (int j = 0; j < 8; ++j)
                pa[g2 * 8 + j] = (__bf16)Pl[wave][mr][g2 * 16 + half * 8 + j];
#pragma unroll
        for (int dt = 0; dt < 4; ++dt) {
            v16bf vfrag;
#pragma unroll
            for (int g2 = 0; g2 < 2; ++g2)
#pragma unroll
                for (int j = 0; j < 8; ++j)
                    vfrag[g2 * 8 + j] =
                        V[base + (size_t)(kvbase + g2 * 16 + half * 8 + j) * D_MODEL +
                          (dt * 16 + mr)];
            acc[dt] = wmma_bf16(pa, vfrag, acc[dt]);
        }
    }
    // ---- epilogue: O = acc / l ----
#pragma unroll
    for (int r = 0; r < 8; ++r) {
        const float inv = 1.0f / rl[r];
        const int q = qbase + half * 8 + r;
#pragma unroll
        for (int dt = 0; dt < 4; ++dt)
            O[base + (size_t)q * D_MODEL + (dt * 16 + mr)] =
                (__bf16)(acc[dt][r] * inv);
    }
}

// ---------------------------------------------------------------------------
// host launcher
// ---------------------------------------------------------------------------
extern "C" void kernel_launch(void* const* d_in, const int* in_sizes, int n_in,
                              void* d_out, int out_size, void* d_ws, size_t ws_size,
                              hipStream_t stream) {
    (void)in_sizes; (void)n_in; (void)out_size; (void)ws_size;
    const float* x   = (const float*)d_in[0];
    const float* Wq  = (const float*)d_in[1];
    const float* bq  = (const float*)d_in[2];
    const float* Wk  = (const float*)d_in[3];
    const float* bk  = (const float*)d_in[4];
    const float* Wv  = (const float*)d_in[5];
    const float* bv  = (const float*)d_in[6];
    const float* Wo  = (const float*)d_in[7];
    const float* bo  = (const float*)d_in[8];
    const float* g1  = (const float*)d_in[9];
    const float* b1  = (const float*)d_in[10];
    const float* g2  = (const float*)d_in[11];
    const float* b2  = (const float*)d_in[12];
    const float* Wup = (const float*)d_in[13];
    const float* bup = (const float*)d_in[14];
    const float* Wdn = (const float*)d_in[15];
    const float* bdn = (const float*)d_in[16];

    char* ws = (char*)d_ws;
    size_t off = 0;
    auto take = [&](size_t bytes) -> void* {
        void* p = ws + off;
        off = (off + bytes + 255) & ~(size_t)255;
        return p;
    };
    __bf16* wq_b  = (__bf16*)take((size_t)D_MODEL * D_MODEL * 2);
    __bf16* wk_b  = (__bf16*)take((size_t)D_MODEL * D_MODEL * 2);
    __bf16* wv_b  = (__bf16*)take((size_t)D_MODEL * D_MODEL * 2);
    __bf16* wo_b  = (__bf16*)take((size_t)D_MODEL * D_MODEL * 2);
    __bf16* wup_b = (__bf16*)take((size_t)D_MODEL * MLP_DIM * 2);
    __bf16* wdn_b = (__bf16*)take((size_t)MLP_DIM * D_MODEL * 2);
    __bf16* xn_b  = (__bf16*)take((size_t)ROWS * D_MODEL * 2);
    __bf16* q_b   = (__bf16*)take((size_t)ROWS * D_MODEL * 2);
    __bf16* k_b   = (__bf16*)take((size_t)ROWS * D_MODEL * 2);
    __bf16* v_b   = (__bf16*)take((size_t)ROWS * D_MODEL * 2);
    __bf16* o_b   = (__bf16*)take((size_t)ROWS * D_MODEL * 2);
    float*  x2    = (float*) take((size_t)ROWS * D_MODEL * 4);
    __bf16* xn2_b = (__bf16*)take((size_t)ROWS * D_MODEL * 2);
    __bf16* h_b   = (__bf16*)take((size_t)ROWS * MLP_DIM * 2);

    // 1) weights fp32 -> bf16
    cvt_f32_bf16_kernel<<<1024, 256, 0, stream>>>(Wq,  wq_b,  D_MODEL * D_MODEL);
    cvt_f32_bf16_kernel<<<1024, 256, 0, stream>>>(Wk,  wk_b,  D_MODEL * D_MODEL);
    cvt_f32_bf16_kernel<<<1024, 256, 0, stream>>>(Wv,  wv_b,  D_MODEL * D_MODEL);
    cvt_f32_bf16_kernel<<<1024, 256, 0, stream>>>(Wo,  wo_b,  D_MODEL * D_MODEL);
    cvt_f32_bf16_kernel<<<2048, 256, 0, stream>>>(Wup, wup_b, D_MODEL * MLP_DIM);
    cvt_f32_bf16_kernel<<<2048, 256, 0, stream>>>(Wdn, wdn_b, MLP_DIM * D_MODEL);

    // 2) LN1 -> bf16
    ln_kernel<<<ROWS, 256, 0, stream>>>(x, g1, b1, xn_b);

    // 3) Q/K/V projections (bf16 out)
    dim3 gD(D_MODEL / BN, ROWS / BM);   // (8, 64)
    gemm_bf16_kernel<0><<<gD, 256, 0, stream>>>(xn_b, wq_b, bq, nullptr, nullptr, q_b,
                                                ROWS, D_MODEL, D_MODEL);
    gemm_bf16_kernel<0><<<gD, 256, 0, stream>>>(xn_b, wk_b, bk, nullptr, nullptr, k_b,
                                                ROWS, D_MODEL, D_MODEL);
    gemm_bf16_kernel<0><<<gD, 256, 0, stream>>>(xn_b, wv_b, bv, nullptr, nullptr, v_b,
                                                ROWS, D_MODEL, D_MODEL);

    // 4) causal flash attention (4096 waves, 4 waves/block)
    attn_kernel<<<(BATCH * NUM_HEADS * (SEQ / 16)) / 4, 128, 0, stream>>>(q_b, k_b, v_b, o_b);

    // 5) output projection + residual -> x2 (fp32)
    gemm_bf16_kernel<0><<<gD, 256, 0, stream>>>(o_b, wo_b, bo, x, x2, nullptr,
                                                ROWS, D_MODEL, D_MODEL);

    // 6) LN2 -> bf16
    ln_kernel<<<ROWS, 256, 0, stream>>>(x2, g2, b2, xn2_b);

    // 7) MLP up + GELU (bf16 out)
    dim3 gU(MLP_DIM / BN, ROWS / BM);   // (32, 64)
    gemm_bf16_kernel<1><<<gU, 256, 0, stream>>>(xn2_b, wup_b, bup, nullptr, nullptr, h_b,
                                                ROWS, MLP_DIM, D_MODEL);

    // 8) MLP down + residual -> out (fp32)
    gemm_bf16_kernel<0><<<gD, 256, 0, stream>>>(h_b, wdn_b, bdn, x2, (float*)d_out, nullptr,
                                                ROWS, D_MODEL, MLP_DIM);
}
